// GATNetV2_7670811591306
// MI455X (gfx1250) — matrix operand order repaired and checked by reading, not compiled, not run on hardware
//
#include <hip/hip_runtime.h>
#include <hip/hip_bf16.h>

typedef __attribute__((ext_vector_type(16))) __bf16 v16bf;
typedef __attribute__((ext_vector_type(8)))  __bf16 v8bf;
typedef __attribute__((ext_vector_type(8)))  float  v8f;

#define NN     10000
#define NEDGE  320000
#define ETOT   (NEDGE + NN)
#define NG     100
#define HS     256      // padded channel stride (250 -> 256)
#define COUT   250
#define K1     336
#define K1P    352      // 336 padded to mult. of 32
#define MAXKP  352
#define TPB    256
#define NSLOPE 0.2f

static __device__ __forceinline__ void atomicMaxF(float* a, float v) {
  // IEEE-monotonic bit trick: int-max for >=0, uint-min for <0.
  if (v >= 0.f) atomicMax((int*)a, __float_as_int(v));
  else          atomicMin((unsigned int*)a, (unsigned int)__float_as_int(v));
}

// ---------------- data prep ----------------
__global__ void k_zero(float* p, int n) {
  int i = blockIdx.x * blockDim.x + threadIdx.x;
  if (i < n) p[i] = 0.f;
}

__global__ void k_cvt_x(const float* __restrict__ x, __bf16* __restrict__ xbf) {
  int i = blockIdx.x * blockDim.x + threadIdx.x;
  if (i >= NN * K1P) return;
  int n = i / K1P, k = i - n * K1P;
  xbf[i] = (k < K1) ? (__bf16)x[n * K1 + k] : (__bf16)0.f;
}

// WT[col][k] = W[k][col], bf16, zero padded. 256 cols x kpad.
__global__ void k_prep_wt(const float* __restrict__ W, __bf16* __restrict__ WT,
                          int K, int kpad) {
  int i = blockIdx.x * blockDim.x + threadIdx.x;
  if (i >= 256 * kpad) return;
  int col = i / kpad, k = i - col * kpad;
  float v = (col < COUT && k < K) ? W[k * COUT + col] : 0.f;
  WT[i] = (__bf16)v;
}

// ---------------- WMMA GEMM: Out[M x 256] = A[M x kpad] * WT^T ----------------
// Full transposed weight panel (<=180KB) staged in the 320KB WGP LDS once per
// block. Each wave computes a 16x64 strip: 4 accumulators, A fragment reused
// across 4 WMMAs per k-step. 625 mtiles * 4 ngroups = 2500 waves.
__global__ __launch_bounds__(TPB)
void k_gemm_wmma(const __bf16* __restrict__ A, int lda,
                 const __bf16* __restrict__ WT,
                 float* __restrict__ Out, int kpad) {
  __shared__ __bf16 sWT[256 * MAXKP];   // 180,224 B of 320 KB WGP LDS
  {
    const v8bf* src = (const v8bf*)WT;
    v8bf* dst = (v8bf*)sWT;
    int total = (256 * kpad) / 8;       // v8bf chunks
    for (int i = threadIdx.x; i < total; i += TPB) dst[i] = src[i];
  }
  __syncthreads();

  int wave = (blockIdx.x * blockDim.x + threadIdx.x) >> 5;
  if (wave >= 625 * 4) return;          // tail block (no barriers past here)
  int lane = threadIdx.x & 31;
  int mt = wave >> 2;                   // 0..624
  int ng = wave & 3;                    // group of 4 ntiles -> cols [ng*64, ng*64+64)
  int r    = lane & 15;
  int half = lane >> 4;                 // 0 or 1

  // A fragment: lane holds row (mt*16+r), K = {16*half .. 16*half+15} contiguous
  const __bf16* ap = A + (size_t)(mt * 16 + r) * lda + half * 16;
  // B fragments from LDS: lane holds col, K = {16*half .. 16*half+15} contiguous
  const __bf16* bl = sWT + (size_t)(ng * 64 + r) * kpad + half * 16;
  const int bstep = 16 * kpad;          // col stride between the 4 tiles

  v8f c0 = {}, c1 = {}, c2 = {}, c3 = {};
  for (int kk = 0; kk < kpad; kk += 32) {
    __builtin_prefetch(ap + kk + 64, 0, 0);
    v8bf alo = *(const v8bf*)(ap + kk);
    v8bf ahi = *(const v8bf*)(ap + kk + 8);
    v16bf a;
#pragma unroll
    for (int i = 0; i < 8; ++i) { a[i] = alo[i]; a[8 + i] = ahi[i]; }
    v16bf b0 = *(const v16bf*)(bl + kk);
    v16bf b1 = *(const v16bf*)(bl + bstep + kk);
    v16bf b2 = *(const v16bf*)(bl + 2 * bstep + kk);
    v16bf b3 = *(const v16bf*)(bl + 3 * bstep + kk);
    c0 = __builtin_amdgcn_wmma_f32_16x16x32_bf16(false, a, false, b0, (short)0, c0, false, false);
    c1 = __builtin_amdgcn_wmma_f32_16x16x32_bf16(false, a, false, b1, (short)0, c1, false, false);
    c2 = __builtin_amdgcn_wmma_f32_16x16x32_bf16(false, a, false, b2, (short)0, c2, false, false);
    c3 = __builtin_amdgcn_wmma_f32_16x16x32_bf16(false, a, false, b3, (short)0, c3, false, false);
  }

  // D layout: VGPR v -> row = mt*16 + 8*half + v, col = ntile*16 + r
  float* orow = Out + (size_t)(mt * 16 + half * 8) * HS + ng * 64 + r;
#pragma unroll
  for (int v = 0; v < 8; ++v) {
    orow[v * HS + 0]  = c0[v];
    orow[v * HS + 16] = c1[v];
    orow[v * HS + 32] = c2[v];
    orow[v * HS + 48] = c3[v];
  }
}

// ---------------- attention coefficients ----------------
__global__ void k_att(const float* __restrict__ h, const float* __restrict__ asrc,
                      const float* __restrict__ adst, float* __restrict__ esrc,
                      float* __restrict__ edst) {
  int i = blockIdx.x * blockDim.x + threadIdx.x;
  if (i >= NN * 2) return;
  int n = i >> 1, hd = i & 1;
  const float* hp = h + (size_t)n * HS + hd * 125;
  const float* as = asrc + hd * 125;
  const float* ad = adst + hd * 125;
  float s0 = 0.f, s1 = 0.f;
  for (int c = 0; c < 125; ++c) { float v = hp[c]; s0 += v * as[c]; s1 += v * ad[c]; }
  esrc[i] = s0; edst[i] = s1;
}

__global__ void k_init_md(float* m, float* denom) {
  int i = blockIdx.x * blockDim.x + threadIdx.x;
  if (i >= NN * 2) return;
  m[i] = -1e30f; denom[i] = 0.f;
}

// ---------------- edge passes ----------------
__global__ void k_edge_max(const int* __restrict__ ei, const float* __restrict__ esrc,
                           const float* __restrict__ edst, float* __restrict__ m) {
  int i = blockIdx.x * blockDim.x + threadIdx.x;
  if (i >= ETOT * 2) return;
  int e = i >> 1, hd = i & 1;
  int s = (e < NEDGE) ? ei[e] : (e - NEDGE);
  int d = (e < NEDGE) ? ei[NEDGE + e] : (e - NEDGE);
  float v = esrc[s * 2 + hd] + edst[d * 2 + hd];
  v = (v > 0.f) ? v : NSLOPE * v;
  atomicMaxF(&m[d * 2 + hd], v);
}

__global__ void k_edge_exp(const int* __restrict__ ei, const float* __restrict__ esrc,
                           const float* __restrict__ edst, const float* __restrict__ m,
                           float* __restrict__ ew, float* __restrict__ denom) {
  int i = blockIdx.x * blockDim.x + threadIdx.x;
  if (i >= ETOT * 2) return;
  int e = i >> 1, hd = i & 1;
  int s = (e < NEDGE) ? ei[e] : (e - NEDGE);
  int d = (e < NEDGE) ? ei[NEDGE + e] : (e - NEDGE);
  float v = esrc[s * 2 + hd] + edst[d * 2 + hd];
  v = (v > 0.f) ? v : NSLOPE * v;
  float w = __expf(v - m[d * 2 + hd]);
  ew[i] = w;
  atomicAdd(&denom[d * 2 + hd], w);
}

__global__ void k_edge_norm(const int* __restrict__ ei, const float* __restrict__ denom,
                            float* __restrict__ ew) {
  int i = blockIdx.x * blockDim.x + threadIdx.x;
  if (i >= ETOT * 2) return;
  int e = i >> 1, hd = i & 1;
  int d = (e < NEDGE) ? ei[NEDGE + e] : (e - NEDGE);
  ew[i] = ew[i] / (denom[d * 2 + hd] + 1e-16f);
}

__global__ void k_edge_scatter(const int* __restrict__ ei, const float* __restrict__ ew,
                               const float* __restrict__ h, float* __restrict__ out) {
  long i = (long)blockIdx.x * blockDim.x + threadIdx.x;
  if (i >= (long)ETOT * COUT) return;
  int e = (int)(i / COUT);
  int c = (int)(i - (long)e * COUT);
  int hd = (c >= 125) ? 1 : 0;
  int s = (e < NEDGE) ? ei[e] : (e - NEDGE);
  int d = (e < NEDGE) ? ei[NEDGE + e] : (e - NEDGE);
  float a = ew[e * 2 + hd];
  atomicAdd(&out[(size_t)d * HS + c], a * h[(size_t)s * HS + c]);
}

// ---------------- layer finalize: relu(out+b) -> f32 act + bf16 copy ----------------
__global__ void k_finalize(float* __restrict__ out, const float* __restrict__ bias,
                           __bf16* __restrict__ hbf) {
  int i = blockIdx.x * blockDim.x + threadIdx.x;
  if (i >= NN * HS) return;
  int c = i & (HS - 1);
  float v = 0.f;
  if (c < COUT) { v = out[i] + bias[c]; v = fmaxf(v, 0.f); }
  out[i] = v;
  hbf[i] = (__bf16)v;
}

// ---------------- pooling ----------------
__global__ void k_pool_sum(const float* __restrict__ act, const int* __restrict__ batch,
                           float* __restrict__ gsum) {
  int i = blockIdx.x * blockDim.x + threadIdx.x;
  if (i >= NN * COUT) return;
  int n = i / COUT, c = i - n * COUT;
  atomicAdd(&gsum[(size_t)batch[n] * HS + c], act[(size_t)n * HS + c]);
}

__global__ void k_pool_cnt(const int* __restrict__ batch, float* __restrict__ gcnt) {
  int n = blockIdx.x * blockDim.x + threadIdx.x;
  if (n >= NN) return;
  atomicAdd(&gcnt[batch[n]], 1.f);
}

__global__ void k_pool_mean(const float* __restrict__ gsum, const float* __restrict__ gcnt,
                            float* __restrict__ g0) {
  int i = blockIdx.x * blockDim.x + threadIdx.x;
  if (i >= NG * COUT) return;
  int g = i / COUT, c = i - g * COUT;
  g0[i] = gsum[(size_t)g * HS + c] / fmaxf(gcnt[g], 1.f);
}

// ---------------- tiny MLP head ----------------
__global__ void k_dense(const float* __restrict__ in, int din,
                        const float* __restrict__ w, const float* __restrict__ b,
                        float* __restrict__ out, int dout, int do_relu) {
  int i = blockIdx.x * blockDim.x + threadIdx.x;
  if (i >= NG * dout) return;
  int g = i / dout, o = i - g * dout;
  float s = b[o];
  const float* ip = in + (size_t)g * din;
  for (int k = 0; k < din; ++k) s += ip[k] * w[k * dout + o];
  out[i] = do_relu ? fmaxf(s, 0.f) : s;
}

// ---------------- host ----------------
static inline int grid_of(long n) { return (int)((n + TPB - 1) / TPB); }

extern "C" void kernel_launch(void* const* d_in, const int* in_sizes, int n_in,
                              void* d_out, int out_size, void* d_ws, size_t ws_size,
                              hipStream_t stream) {
  const float* x     = (const float*)d_in[0];
  const int*   ei    = (const int*)d_in[1];
  const int*   batch = (const int*)d_in[2];
  const float *W[4], *ASRC[4], *ADST[4], *BB[4];
  for (int l = 0; l < 4; ++l) {
    W[l]    = (const float*)d_in[3 + l * 4];
    ASRC[l] = (const float*)d_in[4 + l * 4];
    ADST[l] = (const float*)d_in[5 + l * 4];
    BB[l]   = (const float*)d_in[6 + l * 4];
  }
  const float* lw1 = (const float*)d_in[19]; const float* lb1 = (const float*)d_in[20];
  const float* lw2 = (const float*)d_in[21]; const float* lb2 = (const float*)d_in[22];
  const float* lw3 = (const float*)d_in[23]; const float* lb3 = (const float*)d_in[24];
  const float* lw4 = (const float*)d_in[25]; const float* lb4 = (const float*)d_in[26];
  float* out = (float*)d_out;

  // workspace carve-up (256B aligned)
  char* base = (char*)d_ws;
  size_t off = 0;
  auto carve = [&](size_t bytes) -> char* {
    char* p = base + off;
    off = (off + bytes + 255) & ~(size_t)255;
    return p;
  };
  __bf16* xbf   = (__bf16*)carve((size_t)NN * K1P * 2);
  __bf16* wt    = (__bf16*)carve((size_t)256 * K1P * 2);
  float*  hlin  = (float*)carve((size_t)NN * HS * 4);
  float*  outac = (float*)carve((size_t)NN * HS * 4);
  __bf16* hbf   = (__bf16*)carve((size_t)NN * HS * 2);
  float*  esrc  = (float*)carve((size_t)NN * 2 * 4);
  float*  edst  = (float*)carve((size_t)NN * 2 * 4);
  float*  mmax  = (float*)carve((size_t)NN * 2 * 4);
  float*  denom = (float*)carve((size_t)NN * 2 * 4);
  float*  ew    = (float*)carve((size_t)ETOT * 2 * 4);
  float*  gsum  = (float*)carve((size_t)NG * HS * 4);
  float*  gcnt  = (float*)carve((size_t)NG * 4);
  float*  g0    = (float*)carve((size_t)NG * COUT * 4);
  float*  g1    = (float*)carve((size_t)NG * 200 * 4);
  float*  g2    = (float*)carve((size_t)NG * 100 * 4);
  float*  g3    = (float*)carve((size_t)NG * 100 * 4);

  // x -> bf16 (padded K)
  k_cvt_x<<<grid_of((long)NN * K1P), TPB, 0, stream>>>(x, xbf);

  // 2500 GEMM waves, 8 waves/block -> 313 blocks (tail guarded)
  const int gemm_blocks = (625 * 4 * 32 + TPB - 1) / TPB;

  for (int l = 0; l < 4; ++l) {
    int K    = (l == 0) ? K1 : COUT;
    int kpad = (l == 0) ? K1P : HS;
    const __bf16* A  = (l == 0) ? xbf : hbf;
    int lda          = (l == 0) ? K1P : HS;

    k_prep_wt<<<grid_of((long)256 * kpad), TPB, 0, stream>>>(W[l], wt, K, kpad);
    k_gemm_wmma<<<gemm_blocks, TPB, 0, stream>>>(A, lda, wt, hlin, kpad);

    k_att<<<grid_of((long)NN * 2), TPB, 0, stream>>>(hlin, ASRC[l], ADST[l], esrc, edst);
    k_init_md<<<grid_of((long)NN * 2), TPB, 0, stream>>>(mmax, denom);
    k_zero<<<grid_of((long)NN * HS), TPB, 0, stream>>>(outac, NN * HS);

    k_edge_max<<<grid_of((long)ETOT * 2), TPB, 0, stream>>>(ei, esrc, edst, mmax);
    k_edge_exp<<<grid_of((long)ETOT * 2), TPB, 0, stream>>>(ei, esrc, edst, mmax, ew, denom);
    k_edge_norm<<<grid_of((long)ETOT * 2), TPB, 0, stream>>>(ei, denom, ew);
    k_edge_scatter<<<grid_of((long)ETOT * COUT), TPB, 0, stream>>>(ei, ew, hlin, outac);

    k_finalize<<<grid_of((long)NN * HS), TPB, 0, stream>>>(outac, BB[l], hbf);
  }

  // global mean pool
  k_zero<<<grid_of((long)NG * HS), TPB, 0, stream>>>(gsum, NG * HS);
  k_zero<<<grid_of((long)NG), TPB, 0, stream>>>(gcnt, NG);
  k_pool_sum<<<grid_of((long)NN * COUT), TPB, 0, stream>>>(outac, batch, gsum);
  k_pool_cnt<<<grid_of((long)NN), TPB, 0, stream>>>(batch, gcnt);
  k_pool_mean<<<grid_of((long)NG * COUT), TPB, 0, stream>>>(gsum, gcnt, g0);

  // MLP head
  k_dense<<<grid_of((long)NG * 200), TPB, 0, stream>>>(g0, 250, lw1, lb1, g1, 200, 1);
  k_dense<<<grid_of((long)NG * 100), TPB, 0, stream>>>(g1, 200, lw2, lb2, g2, 100, 1);
  k_dense<<<grid_of((long)NG * 100), TPB, 0, stream>>>(g2, 100, lw3, lb3, g3, 100, 1);
  k_dense<<<grid_of((long)NG * 29),  TPB, 0, stream>>>(g3, 100, lw4, lb4, out, 29, 0);
}
